// MambaAttentionHead_77412490543273
// MI455X (gfx1250) — matrix operand order repaired and checked by reading, not compiled, not run on hardware
//
#include <hip/hip_runtime.h>
#include <hip/hip_bf16.h>
#include <math.h>

typedef _Float16 h16_t;
typedef __attribute__((ext_vector_type(8)))  _Float16 v8h;
typedef __attribute__((ext_vector_type(16))) _Float16 v16h;
typedef __attribute__((ext_vector_type(8)))  float    v8f;

#define E_DIM   256
#define P_Q     10
#define H_ATT   4
#define FFN_DIM 512
#define DIN     64
#define NFEAT   15
#define BATCH   4
#define SEQ     1024
#define TOK     (BATCH*SEQ)   /* 4096 */
#define DSTATE  64
#define HEADDIM 64
#define DINNER  512
#define MHEADS  8
#define CONVCH  640
#define DPROJ   1160
#define EPSV    1e-5f

__device__ __forceinline__ float geluf(float x) {
    return 0.5f * x * (1.f + erff(x * 0.70710678118654752f));
}
__device__ __forceinline__ float siluf(float x) {
    return x / (1.f + expf(-x));
}
__device__ __forceinline__ v8f zero8f() {
    v8f v;
#pragma unroll
    for (int i = 0; i < 8; ++i) v[i] = 0.f;
    return v;
}
__device__ __forceinline__ v8h zero8h() {
    v8h v;
#pragma unroll
    for (int i = 0; i < 8; ++i) v[i] = (h16_t)0.f;
    return v;
}
__device__ __forceinline__ v16h zero16h() {
    v16h v;
#pragma unroll
    for (int i = 0; i < 16; ++i) v[i] = (h16_t)0.f;
    return v;
}
__device__ __forceinline__ v16h cat8h(v8h lo, v8h hi) {
    v16h a;
#pragma unroll
    for (int i = 0; i < 8; ++i) { a[i] = lo[i]; a[i + 8] = hi[i]; }
    return a;
}

// ---------------------------------------------------------------------------
// Generic WMMA GEMM:  C[M,N] = act( A[M,K] * B[N,K]^T + bias + add )
// A,B are f16 row-major (K contiguous).  K must be a multiple of 32.
// Block = 128 threads = 4 waves; each wave owns one 16-row x 16-col tile.
// Fast path (full tile, wave-uniform): pure loads + v_wmma, no EXEC churn.
// Edge path: clamped in-bounds loads + per-lane v_cndmask zeroing (no
// divergent branches around the WMMA, keeping EXEC all-ones as required).
// Batched via gridDim.z with element strides sA/sB/sC/sC16/sAdd.
// act: 0=none 1=gelu 2=sigmoid
// ---------------------------------------------------------------------------
__global__ void k_gemm(const h16_t* __restrict__ A, int lda, long long sA,
                       const h16_t* __restrict__ Bw, int ldb, long long sB,
                       float* __restrict__ C, int ldc, long long sC,
                       h16_t* __restrict__ C16, int ldc16, long long sC16,
                       const float* __restrict__ bias,
                       const float* __restrict__ addsrc, int ldadd, long long sAdd,
                       int M, int N, int K, int act)
{
    const int lane = threadIdx.x & 31;
    const int wave = threadIdx.x >> 5;
    const int m0 = blockIdx.x * 64 + wave * 16;
    const int n0 = blockIdx.y * 16;
    const long long z = blockIdx.z;

    const int kh = lane >> 4;                 // 0 or 1 (K half-split)
    const int rA = m0 + (lane & 15);
    const int cB = n0 + (lane & 15);
    const bool okA = (rA < M);
    const bool okB = (cB < N);
    // clamped, always-in-bounds per-lane base pointers (contiguous k order, k = k0 + 16*kh + s):
    // A lane: row r, halves 0..7 = A[r][k0+kh*16..], halves 8..15 = +8
    // B lane: col c, halves 0..15 = B[c][k0+kh*16..]
    const h16_t* pA = A + z * sA + (long long)(okA ? rA : (M - 1)) * lda + kh * 16;
    const h16_t* pB = Bw + z * sB + (long long)(okB ? cB : (N - 1)) * ldb + kh * 16;

    v8f acc = zero8f();
    if ((m0 + 16 <= M) & (n0 + 16 <= N)) {
        // interior tile: unconditional streaming loads
        for (int k0 = 0; k0 < K; k0 += 32) {
            const v8h  lo = *(const v8h*)(pA + k0);
            const v8h  hi = *(const v8h*)(pA + k0 + 8);
            const v16h bf = *(const v16h*)(pB + k0);
            __builtin_prefetch(pA + k0 + 32, 0, 1);   // global_prefetch_b8
            __builtin_prefetch(pB + k0 + 32, 0, 1);
            acc = __builtin_amdgcn_wmma_f32_16x16x32_f16(
                false, cat8h(lo, hi), false, bf, (short)0, acc, false, false);
        }
    } else {
        // edge tile: in-bounds loads, lane-masked via selects (no EXEC branch)
        for (int k0 = 0; k0 < K; k0 += 32) {
            v8h  lo = *(const v8h*)(pA + k0);
            v8h  hi = *(const v8h*)(pA + k0 + 8);
            v16h bf = *(const v16h*)(pB + k0);
            lo = okA ? lo : zero8h();
            hi = okA ? hi : zero8h();
            bf = okB ? bf : zero16h();
            acc = __builtin_amdgcn_wmma_f32_16x16x32_f16(
                false, cat8h(lo, hi), false, bf, (short)0, acc, false, false);
        }
    }

    // D layout: VGPR v, lanes 0-15 -> row m0+v, lanes 16-31 -> row m0+v+8
    const int col = n0 + (lane & 15);
    const int rbase = m0 + (lane >> 4) * 8;
    if (col < N) {
        const float bv = bias ? bias[col] : 0.f;
#pragma unroll
        for (int v = 0; v < 8; ++v) {
            const int row = rbase + v;
            if (row < M) {
                float val = acc[v] + bv;
                if (addsrc) val += addsrc[z * sAdd + (long long)row * ldadd + col];
                if (act == 1)      val = geluf(val);
                else if (act == 2) val = 1.f / (1.f + expf(-val));
                if (C)   C[z * sC + (long long)row * ldc + col] = val;
                if (C16) C16[z * sC16 + (long long)row * ldc16 + col] = (h16_t)val;
            }
        }
    }
}

// f32 -> f16 convert
__global__ void k_cvt(const float* __restrict__ s, h16_t* __restrict__ d, int n)
{
    int i = blockIdx.x * blockDim.x + threadIdx.x;
    if (i < n) d[i] = (h16_t)s[i];
}

// softmax of the 15 feature weights (single thread; trivial)
__global__ void k_wsm(const float* __restrict__ w, float* __restrict__ o)
{
    if (threadIdx.x == 0 && blockIdx.x == 0) {
        float m = -1e30f;
        for (int i = 0; i < NFEAT; ++i) m = fmaxf(m, w[i]);
        float e[NFEAT]; float s = 0.f;
        for (int i = 0; i < NFEAT; ++i) { e[i] = expf(w[i] - m); s += e[i]; }
        for (int i = 0; i < NFEAT; ++i) o[i] = e[i] / s;
    }
}

// weighted feature average + input LayerNorm -> f16 [TOK, DIN]
__global__ void k_feat_in(const float* __restrict__ F, const float* __restrict__ sw,
                          const float* __restrict__ g, const float* __restrict__ b,
                          h16_t* __restrict__ o)
{
    __shared__ float sm[DIN];
    const int t = blockIdx.x, d = threadIdx.x;
    float a = 0.f;
    for (int n = 0; n < NFEAT; ++n)
        a += sw[n] * F[((long long)n * TOK + t) * DIN + d];
    sm[d] = a; __syncthreads();
    for (int s = 32; s > 0; s >>= 1) { if (d < s) sm[d] += sm[d + s]; __syncthreads(); }
    const float mean = sm[0] / (float)DIN; __syncthreads();
    const float dv = a - mean;
    sm[d] = dv * dv; __syncthreads();
    for (int s = 32; s > 0; s >>= 1) { if (d < s) sm[d] += sm[d + s]; __syncthreads(); }
    const float var = sm[0] / (float)DIN;
    o[(long long)t * DIN + d] = (h16_t)(dv * rsqrtf(var + EPSV) * g[d] + b[d]);
}

// per-row LayerNorm -> f16 (blockDim = C, power of two)
__global__ void k_ln16(const float* __restrict__ x, const float* __restrict__ g,
                       const float* __restrict__ b, h16_t* __restrict__ o, int C)
{
    extern __shared__ float sm[];
    const int r = blockIdx.x, c = threadIdx.x;
    const float v = x[(long long)r * C + c];
    sm[c] = v; __syncthreads();
    for (int s = C >> 1; s > 0; s >>= 1) { if (c < s) sm[c] += sm[c + s]; __syncthreads(); }
    const float mean = sm[0] / (float)C; __syncthreads();
    const float dv = v - mean;
    sm[c] = dv * dv; __syncthreads();
    for (int s = C >> 1; s > 0; s >>= 1) { if (c < s) sm[c] += sm[c + s]; __syncthreads(); }
    const float var = sm[0] / (float)C;
    o[(long long)r * C + c] = (h16_t)(dv * rsqrtf(var + EPSV) * g[c] + b[c]);
}

// per-row RMSNorm -> optional f16 / f32 out (blockDim = C, power of two)
__global__ void k_rms(const float* __restrict__ x, const float* __restrict__ w,
                      h16_t* __restrict__ o16, float* __restrict__ o32, int C)
{
    extern __shared__ float sm[];
    const int r = blockIdx.x, c = threadIdx.x;
    const float v = x[(long long)r * C + c];
    sm[c] = v * v; __syncthreads();
    for (int s = C >> 1; s > 0; s >>= 1) { if (c < s) sm[c] += sm[c + s]; __syncthreads(); }
    const float inv = rsqrtf(sm[0] / (float)C + EPSV);
    const float out = v * inv * w[c];
    if (o16) o16[(long long)r * C + c] = (h16_t)out;
    if (o32) o32[(long long)r * C + c] = out;
}

// depthwise causal conv (k=4) over xBC channels + SiLU
__global__ void k_conv(const float* __restrict__ zx, const float* __restrict__ cw,
                       const float* __restrict__ cb, float* __restrict__ out)
{
    const int t = blockIdx.x, c = threadIdx.x;
    const int b = t >> 10, s = t & 1023;
    float acc = cb[c];
#pragma unroll
    for (int j = 0; j < 4; ++j) {
        const int ss = s - 3 + j;
        if (ss >= 0)
            acc += cw[c * 4 + j] * zx[((long long)(b * SEQ + ss)) * DPROJ + DINNER + c];
    }
    out[(long long)t * CONVCH + c] = siluf(acc);
}

// dt = softplus(raw + bias), dA = exp(-exp(A_log)*dt)
__global__ void k_dt(const float* __restrict__ zx, const float* __restrict__ dtbias,
                     const float* __restrict__ alog,
                     float* __restrict__ dtb, float* __restrict__ dab, int n)
{
    const int i = blockIdx.x * blockDim.x + threadIdx.x;
    if (i >= n) return;
    const int t = i >> 3, h = i & 7;
    const float raw = zx[(long long)t * DPROJ + (DPROJ - MHEADS) + h] + dtbias[h];
    const float sp = (raw > 20.f) ? raw : log1pf(expf(raw));
    dtb[i] = sp;
    dab[i] = expf(-expf(alog[h]) * sp);
}

// sequential selective-state scan: one block per (batch, head); state in regs
__global__ void k_scan(const float* __restrict__ xbc, const float* __restrict__ dtb,
                       const float* __restrict__ dab, float* __restrict__ y)
{
    const int b = blockIdx.x >> 3;
    const int h = blockIdx.x & 7;
    const int tid = threadIdx.x;
    const int d = tid & 63;
    const int ng = tid >> 6;          // 4 groups x 16 states each
    __shared__ float sB[DSTATE], sC[DSTATE], red[256];
    float hs[16];
#pragma unroll
    for (int i = 0; i < 16; ++i) hs[i] = 0.f;

    for (int t = 0; t < SEQ; ++t) {
        const int tt = b * SEQ + t;
        const float* row = xbc + (long long)tt * CONVCH;
        if (tid < 64)       sB[tid]      = row[DINNER + tid];
        else if (tid < 128) sC[tid - 64] = row[DINNER + DSTATE + (tid - 64)];
        __syncthreads();
        const float dA = dab[tt * MHEADS + h];
        const float dt = dtb[tt * MHEADS + h];
        const float coef = dt * row[h * HEADDIM + d];
        float part = 0.f;
#pragma unroll
        for (int i = 0; i < 16; ++i) {
            const int n = ng * 16 + i;
            hs[i] = dA * hs[i] + coef * sB[n];
            part += hs[i] * sC[n];
        }
        red[tid] = part;
        __syncthreads();
        if (tid < 64)
            y[(long long)tt * DINNER + h * HEADDIM + d] =
                red[d] + red[64 + d] + red[128 + d] + red[192 + d];
        __syncthreads();
    }
}

// (y + D*x) * silu(z), RMSNorm over 512 -> f16
__global__ void k_gate(const float* __restrict__ yb, const float* __restrict__ xbc,
                       const float* __restrict__ zx, const float* __restrict__ Dp,
                       const float* __restrict__ nw, h16_t* __restrict__ out)
{
    __shared__ float red[DINNER];
    const int t = blockIdx.x, c = threadIdx.x;
    const float xv = xbc[(long long)t * CONVCH + c];
    const float yv = yb[(long long)t * DINNER + c] + Dp[c >> 6] * xv;
    const float zv = zx[(long long)t * DPROJ + c];
    const float g = yv * siluf(zv);
    red[c] = g * g; __syncthreads();
    for (int s = DINNER >> 1; s > 0; s >>= 1) { if (c < s) red[c] += red[c + s]; __syncthreads(); }
    const float inv = rsqrtf(red[0] / (float)DINNER + EPSV);
    out[(long long)t * DINNER + c] = (h16_t)(g * inv * nw[c]);
}

// proto broadcast / +pos  (rp layout (B,P,E), row = b*10+p)
__global__ void k_proto(const float* __restrict__ pe, float* __restrict__ rp)
{
    const int r = blockIdx.x, c = threadIdx.x;
    rp[r * E_DIM + c] = pe[(r % P_Q) * E_DIM + c];
}
__global__ void k_addpos(const float* __restrict__ rp, const float* __restrict__ pos,
                         float* __restrict__ q)
{
    const int r = blockIdx.x, c = threadIdx.x;
    q[r * E_DIM + c] = rp[r * E_DIM + c] + pos[(r % P_Q) * E_DIM + c];
}

// cross-attention, one (b,h) per block; bias = log(mask+1e-6); padding all-true
__global__ void k_attn_ca(const float* __restrict__ Qb, const float* __restrict__ Kb,
                          const float* __restrict__ Vb, const float* __restrict__ maskp,
                          h16_t* __restrict__ O16)
{
    const int b = blockIdx.x >> 2;
    const int h = blockIdx.x & 3;
    __shared__ float sS[P_Q * SEQ];
    __shared__ float red[256];
    const int tid = threadIdx.x;

    for (int idx = tid; idx < P_Q * SEQ; idx += 256) {
        const int q = idx >> 10, k = idx & 1023;
        const float* qp = Qb + (long long)(b * P_Q + q) * E_DIM + h * 64;
        const float* kp = Kb + (long long)(b * SEQ + k) * E_DIM + h * 64;
        float s = 0.f;
#pragma unroll
        for (int d = 0; d < 64; ++d) s += qp[d] * kp[d];
        sS[idx] = s * 0.125f + logf(maskp[(long long)(b * SEQ + k) * P_Q + q] + 1e-6f);
    }
    __syncthreads();

    for (int q = 0; q < P_Q; ++q) {
        float m = -1e30f;
        for (int k = tid; k < SEQ; k += 256) m = fmaxf(m, sS[q * SEQ + k]);
        red[tid] = m; __syncthreads();
        for (int s = 128; s > 0; s >>= 1) { if (tid < s) red[tid] = fmaxf(red[tid], red[tid + s]); __syncthreads(); }
        m = red[0]; __syncthreads();
        float sum = 0.f;
        for (int k = tid; k < SEQ; k += 256) {
            const float e = expf(sS[q * SEQ + k] - m);
            sS[q * SEQ + k] = e; sum += e;
        }
        red[tid] = sum; __syncthreads();
        for (int s = 128; s > 0; s >>= 1) { if (tid < s) red[tid] += red[tid + s]; __syncthreads(); }
        const float inv = 1.f / red[0]; __syncthreads();
        for (int k = tid; k < SEQ; k += 256) sS[q * SEQ + k] *= inv;
        __syncthreads();
    }

    for (int idx = tid; idx < P_Q * 64; idx += 256) {
        const int q = idx >> 6, d = idx & 63;
        float acc = 0.f;
        for (int k = 0; k < SEQ; ++k)
            acc += sS[q * SEQ + k] * Vb[(long long)(b * SEQ + k) * E_DIM + h * 64 + d];
        O16[(long long)(b * P_Q + q) * E_DIM + h * 64 + d] = (h16_t)acc;
    }
}

// self-attention among the 10 protos, one (b,h) per block
__global__ void k_attn_sa(const float* __restrict__ QKV, h16_t* __restrict__ O16)
{
    const int b = blockIdx.x >> 2;
    const int h = blockIdx.x & 3;
    __shared__ float sS[P_Q * P_Q];
    const int tid = threadIdx.x;

    for (int idx = tid; idx < P_Q * P_Q; idx += 128) {
        const int q = idx / P_Q, k = idx % P_Q;
        const float* qp = QKV + (long long)(b * P_Q + q) * 768 + h * 64;
        const float* kp = QKV + (long long)(b * P_Q + k) * 768 + 256 + h * 64;
        float s = 0.f;
#pragma unroll
        for (int d = 0; d < 64; ++d) s += qp[d] * kp[d];
        sS[idx] = s * 0.125f;
    }
    __syncthreads();
    if (tid < P_Q) {
        float m = -1e30f;
        for (int k = 0; k < P_Q; ++k) m = fmaxf(m, sS[tid * P_Q + k]);
        float sum = 0.f;
        for (int k = 0; k < P_Q; ++k) { const float e = expf(sS[tid * P_Q + k] - m); sS[tid * P_Q + k] = e; sum += e; }
        const float inv = 1.f / sum;
        for (int k = 0; k < P_Q; ++k) sS[tid * P_Q + k] *= inv;
    }
    __syncthreads();
    for (int idx = tid; idx < P_Q * 64; idx += 128) {
        const int q = idx >> 6, d = idx & 63;
        float acc = 0.f;
        for (int k = 0; k < P_Q; ++k)
            acc += sS[q * P_Q + k] * QKV[(long long)(b * P_Q + k) * 768 + 512 + h * 64 + d];
        O16[(long long)(b * P_Q + q) * E_DIM + h * 64 + d] = (h16_t)acc;
    }
}

// softmax over 2 logits for the class head (40 rows)
__global__ void k_softmax2(const float* __restrict__ lg, float* __restrict__ o)
{
    const int r = blockIdx.x * blockDim.x + threadIdx.x;
    if (r < BATCH * P_Q) {
        const float a = lg[2 * r], b = lg[2 * r + 1];
        const float m = fmaxf(a, b);
        const float ea = expf(a - m), eb = expf(b - m);
        const float s = ea + eb;
        o[2 * r] = ea / s; o[2 * r + 1] = eb / s;
    }
}

// ---------------------------------------------------------------------------
extern "C" void kernel_launch(void* const* d_in, const int* in_sizes, int n_in,
                              void* d_out, int out_size, void* d_ws, size_t ws_size,
                              hipStream_t stream)
{
    (void)in_sizes; (void)n_in; (void)out_size; (void)ws_size;
    // ---- input index map (jax pytree: dict keys sorted, lists in order) ----
    enum {
        I_FEATURE = 0, I_PADMASK = 1,
        I_CLS_B1 = 2, I_CLS_B2, I_CLS_LNB, I_CLS_LNG, I_CLS_W1, I_CLS_W2,
        I_FINAL_RMS = 8, I_IN_B = 9, I_IN_LNB = 10, I_IN_LNG = 11, I_IN_W = 12,
        I_MAMBA = 13,   // 9/layer: A_log,D,conv_b,conv_w,dt_bias,in_w,norm_w,out_w,rms_w
        I_MASK_B1 = 40, I_MASK_B2, I_MASK_LNB, I_MASK_LNG, I_MASK_W1, I_MASK_W2,
        I_NOI_B1 = 46, I_NOI_B2, I_NOI_LNB, I_NOI_LNG, I_NOI_W1, I_NOI_W2,
        I_OUT_B1 = 52, I_OUT_B2, I_OUT_LNB, I_OUT_LNG, I_OUT_W1, I_OUT_W2,
        I_PID_B1 = 58, I_PID_B2, I_PID_LNB, I_PID_LNG, I_PID_W1, I_PID_W2,
        I_POS = 64, I_PROTO = 65,
        I_REF = 66,     // 18/layer: ca_bi,ca_bo,ca_ln_b,ca_ln_g,ca_wi,ca_wo,
                        //           ffn_b1,ffn_b2,ffn_ln_b,ffn_ln_g,ffn_w1,ffn_w2,
                        //           sa_bi,sa_bo,sa_ln_b,sa_ln_g,sa_wi,sa_wo
        I_TRK_B1 = 102, I_TRK_B2, I_TRK_LNB, I_TRK_LNG, I_TRK_W1, I_TRK_W2,
        I_WAVG = 108
    };
    enum { M_ALOG = 0, M_D, M_CONVB, M_CONVW, M_DTB, M_INW, M_NORMW, M_OUTW, M_RMSW };
    enum { R_CA_BI = 0, R_CA_BO, R_CA_LNB, R_CA_LNG, R_CA_WI, R_CA_WO,
           R_FFN_B1, R_FFN_B2, R_FFN_LNB, R_FFN_LNG, R_FFN_W1, R_FFN_W2,
           R_SA_BI, R_SA_BO, R_SA_LNB, R_SA_LNG, R_SA_WI, R_SA_WO };

    auto A  = [&](int i) { return (const float*)d_in[i]; };
    auto MB = [&](int l, int f) { return (const float*)d_in[I_MAMBA + 9 * l + f]; };
    auto RF = [&](int l, int f) { return (const float*)d_in[I_REF + 18 * l + f]; };
    const float* feat = A(I_FEATURE);
    float* out = (float*)d_out;
    // output regions (flat concat): cls(80) mask(40960) noise(8192) track(160) pid(240)
    float* o_cls   = out;
    float* o_mask  = out + 80;
    float* o_noise = out + 80 + 40960;
    float* o_track = out + 80 + 40960 + 8192;
    float* o_pid   = out + 80 + 40960 + 8192 + 160;

    // ---- workspace bump allocator ----
    char* bp = (char*)d_ws;
    auto alloc = [&](size_t bytes) -> char* {
        char* r = bp; bp += (bytes + 255) & ~(size_t)255; return r;
    };
    float* sw    = (float*)alloc(64);
    h16_t* xin16 = (h16_t*)alloc((size_t)TOK * DIN * 2);
    float* x     = (float*)alloc((size_t)TOK * E_DIM * 4);
    h16_t* h16   = (h16_t*)alloc((size_t)TOK * E_DIM * 2);
    float* zx    = (float*)alloc((size_t)TOK * DPROJ * 4);
    float* xBCa  = (float*)alloc((size_t)TOK * CONVCH * 4);
    float* dtb   = (float*)alloc((size_t)TOK * MHEADS * 4);
    float* dab   = (float*)alloc((size_t)TOK * MHEADS * 4);
    float* ybuf  = (float*)alloc((size_t)TOK * DINNER * 4);
    h16_t* y16   = (h16_t*)alloc((size_t)TOK * DINNER * 2);
    float* xf    = (float*)alloc((size_t)TOK * E_DIM * 4);
    float* tpf   = (float*)alloc((size_t)TOK * E_DIM * 4);
    h16_t* tp16  = (h16_t*)alloc((size_t)TOK * E_DIM * 2);
    float* maskp = (float*)alloc((size_t)TOK * P_Q * 4);
    float* rp    = (float*)alloc(40 * E_DIM * 4);
    float* qbuf  = (float*)alloc(40 * E_DIM * 4);
    float* Qb    = (float*)alloc(40 * E_DIM * 4);
    float* QKV   = (float*)alloc(40 * 768 * 4);
    h16_t* xn16  = (h16_t*)alloc(40 * E_DIM * 2);
    h16_t* lnp16 = (h16_t*)alloc(40 * E_DIM * 2);
    h16_t* Ob16  = (h16_t*)alloc(40 * E_DIM * 2);
    h16_t* mp16  = (h16_t*)alloc(40 * E_DIM * 2);
    h16_t* hid40 = (h16_t*)alloc(40 * 512 * 2);
    float* logit = (float*)alloc(40 * 2 * 4);
    // aliases into big buffers that are dead after the Mamba stack:
    float* Kb    = zx;                              // [TOK, E] f32
    float* Vb    = zx + (size_t)TOK * E_DIM;        // [TOK, E] f32
    h16_t* ln16  = (h16_t*)xBCa;                    // [TOK, E] f16
    h16_t* hid16 = ln16 + (size_t)TOK * E_DIM;      // [TOK, E] f16
    // f16 weights
    h16_t* w_in = (h16_t*)alloc((size_t)E_DIM * DIN * 2);
    h16_t* w_min[3]; h16_t* w_mout[3];
    for (int l = 0; l < 3; ++l) {
        w_min[l]  = (h16_t*)alloc((size_t)DPROJ * E_DIM * 2);
        w_mout[l] = (h16_t*)alloc((size_t)E_DIM * DINNER * 2);
    }
    h16_t* w_out1 = (h16_t*)alloc((size_t)E_DIM * E_DIM * 2);
    h16_t* w_out2 = (h16_t*)alloc((size_t)E_DIM * E_DIM * 2);
    h16_t* w_noi1 = (h16_t*)alloc((size_t)E_DIM * E_DIM * 2);
    h16_t* w_noi2 = (h16_t*)alloc((size_t)2 * E_DIM * 2);
    h16_t* w_cls1 = (h16_t*)alloc((size_t)E_DIM * E_DIM * 2);
    h16_t* w_cls2 = (h16_t*)alloc((size_t)2 * E_DIM * 2);
    h16_t* w_msk1 = (h16_t*)alloc((size_t)E_DIM * E_DIM * 2);
    h16_t* w_msk2 = (h16_t*)alloc((size_t)E_DIM * E_DIM * 2);
    h16_t* w_trk1 = (h16_t*)alloc((size_t)E_DIM * E_DIM * 2);
    h16_t* w_trk2 = (h16_t*)alloc((size_t)4 * E_DIM * 2);
    h16_t* w_pid1 = (h16_t*)alloc((size_t)E_DIM * E_DIM * 2);
    h16_t* w_pid2 = (h16_t*)alloc((size_t)6 * E_DIM * 2);
    h16_t* w_cawi[2]; h16_t* w_cawo[2]; h16_t* w_sawi[2]; h16_t* w_sawo[2];
    h16_t* w_fw1[2];  h16_t* w_fw2[2];
    for (int l = 0; l < 2; ++l) {
        w_cawi[l] = (h16_t*)alloc((size_t)768 * E_DIM * 2);
        w_cawo[l] = (h16_t*)alloc((size_t)E_DIM * E_DIM * 2);
        w_sawi[l] = (h16_t*)alloc((size_t)768 * E_DIM * 2);
        w_sawo[l] = (h16_t*)alloc((size_t)E_DIM * E_DIM * 2);
        w_fw1[l]  = (h16_t*)alloc((size_t)FFN_DIM * E_DIM * 2);
        w_fw2[l]  = (h16_t*)alloc((size_t)E_DIM * FFN_DIM * 2);
    }

    auto cvt = [&](const float* s, h16_t* d, int n) {
        k_cvt<<<(n + 255) / 256, 256, 0, stream>>>(s, d, n);
    };
    auto gemm = [&](const h16_t* Aa, int lda, long long sA,
                    const h16_t* Bw, int ldb, long long sB,
                    float* C, int ldc, long long sC,
                    h16_t* C16, int ldc16, long long sC16,
                    const float* bias, const float* add, int ldadd, long long sAdd,
                    int M, int N, int K, int act, int batch) {
        dim3 g((M + 63) / 64, (N + 15) / 16, batch);
        k_gemm<<<g, 128, 0, stream>>>(Aa, lda, sA, Bw, ldb, sB, C, ldc, sC,
                                      C16, ldc16, sC16, bias, add, ldadd, sAdd,
                                      M, N, K, act);
    };
    auto ln = [&](const float* xi, const float* g, const float* b, h16_t* o,
                  int rows, int C) {
        k_ln16<<<rows, C, C * 4, stream>>>(xi, g, b, o, C);
    };

    // ---- convert weights to f16 ----
    cvt(A(I_IN_W), w_in, E_DIM * DIN);
    for (int l = 0; l < 3; ++l) {
        cvt(MB(l, M_INW),  w_min[l],  DPROJ * E_DIM);
        cvt(MB(l, M_OUTW), w_mout[l], E_DIM * DINNER);
    }
    cvt(A(I_OUT_W1), w_out1, E_DIM * E_DIM); cvt(A(I_OUT_W2), w_out2, E_DIM * E_DIM);
    cvt(A(I_NOI_W1), w_noi1, E_DIM * E_DIM); cvt(A(I_NOI_W2), w_noi2, 2 * E_DIM);
    cvt(A(I_CLS_W1), w_cls1, E_DIM * E_DIM); cvt(A(I_CLS_W2), w_cls2, 2 * E_DIM);
    cvt(A(I_MASK_W1), w_msk1, E_DIM * E_DIM); cvt(A(I_MASK_W2), w_msk2, E_DIM * E_DIM);
    cvt(A(I_TRK_W1), w_trk1, E_DIM * E_DIM); cvt(A(I_TRK_W2), w_trk2, 4 * E_DIM);
    cvt(A(I_PID_W1), w_pid1, E_DIM * E_DIM); cvt(A(I_PID_W2), w_pid2, 6 * E_DIM);
    for (int l = 0; l < 2; ++l) {
        cvt(RF(l, R_CA_WI), w_cawi[l], 768 * E_DIM);
        cvt(RF(l, R_CA_WO), w_cawo[l], E_DIM * E_DIM);
        cvt(RF(l, R_SA_WI), w_sawi[l], 768 * E_DIM);
        cvt(RF(l, R_SA_WO), w_sawo[l], E_DIM * E_DIM);
        cvt(RF(l, R_FFN_W1), w_fw1[l], FFN_DIM * E_DIM);
        cvt(RF(l, R_FFN_W2), w_fw2[l], E_DIM * FFN_DIM);
    }

    // ---- trunk: feature avg + input LN + projection ----
    k_wsm<<<1, 32, 0, stream>>>(A(I_WAVG), sw);
    k_feat_in<<<TOK, DIN, 0, stream>>>(feat, sw, A(I_IN_LNG), A(I_IN_LNB), xin16);
    gemm(xin16, DIN, 0, w_in, DIN, 0, x, E_DIM, 0, nullptr, 0, 0,
         A(I_IN_B), nullptr, 0, 0, TOK, E_DIM, DIN, 0, 1);

    // ---- Mamba2 blocks ----
    for (int l = 0; l < 3; ++l) {
        k_rms<<<TOK, E_DIM, E_DIM * 4, stream>>>(x, MB(l, M_RMSW), h16, nullptr, E_DIM);
        gemm(h16, E_DIM, 0, w_min[l], E_DIM, 0, zx, DPROJ, 0, nullptr, 0, 0,
             nullptr, nullptr, 0, 0, TOK, DPROJ, E_DIM, 0, 1);
        k_conv<<<TOK, CONVCH, 0, stream>>>(zx, MB(l, M_CONVW), MB(l, M_CONVB), xBCa);
        k_dt<<<(TOK * MHEADS + 255) / 256, 256, 0, stream>>>(
            zx, MB(l, M_DTB), MB(l, M_ALOG), dtb, dab, TOK * MHEADS);
        k_scan<<<BATCH * MHEADS, 256, 0, stream>>>(xBCa, dtb, dab, ybuf);
        k_gate<<<TOK, DINNER, 0, stream>>>(ybuf, xBCa, zx, MB(l, M_D),
                                           MB(l, M_NORMW), y16);
        gemm(y16, DINNER, 0, w_mout[l], DINNER, 0, x, E_DIM, 0, nullptr, 0, 0,
             nullptr, x, E_DIM, 0, TOK, E_DIM, DINNER, 0, 1);   // += residual
    }
    k_rms<<<TOK, E_DIM, E_DIM * 4, stream>>>(x, A(I_FINAL_RMS), nullptr, xf, E_DIM);

    // ---- out_head -> tp ;  noise_head -> d_out ----
    ln(xf, A(I_OUT_LNG), A(I_OUT_LNB), ln16, TOK, E_DIM);
    gemm(ln16, E_DIM, 0, w_out1, E_DIM, 0, nullptr, 0, 0, hid16, E_DIM, 0,
         A(I_OUT_B1), nullptr, 0, 0, TOK, E_DIM, E_DIM, 1, 1);
    gemm(hid16, E_DIM, 0, w_out2, E_DIM, 0, tpf, E_DIM, 0, tp16, E_DIM, 0,
         A(I_OUT_B2), nullptr, 0, 0, TOK, E_DIM, E_DIM, 0, 1);
    ln(xf, A(I_NOI_LNG), A(I_NOI_LNB), ln16, TOK, E_DIM);
    gemm(ln16, E_DIM, 0, w_noi1, E_DIM, 0, nullptr, 0, 0, hid16, E_DIM, 0,
         A(I_NOI_B1), nullptr, 0, 0, TOK, E_DIM, E_DIM, 1, 1);
    gemm(hid16, E_DIM, 0, w_noi2, E_DIM, 0, o_noise, 2, 0, nullptr, 0, 0,
         A(I_NOI_B2), nullptr, 0, 0, TOK, 2, E_DIM, 0, 1);

    // ---- mask-prediction helper (mask_head MLP + batched sigmoid einsum) ----
    auto make_mask = [&](float* dst) {
        ln(rp, A(I_MASK_LNG), A(I_MASK_LNB), lnp16, 40, E_DIM);
        gemm(lnp16, E_DIM, 0, w_msk1, E_DIM, 0, nullptr, 0, 0, hid40, E_DIM, 0,
             A(I_MASK_B1), nullptr, 0, 0, 40, E_DIM, E_DIM, 1, 1);
        gemm(hid40, E_DIM, 0, w_msk2, E_DIM, 0, nullptr, 0, 0, mp16, E_DIM, 0,
             A(I_MASK_B2), nullptr, 0, 0, 40, E_DIM, E_DIM, 0, 1);
        // mask[b,s,p] = sigmoid( tp[b,s,:] . mp[b,p,:] )   (batched over b)
        gemm(tp16, E_DIM, (long long)SEQ * E_DIM,
             mp16, E_DIM, (long long)P_Q * E_DIM,
             dst, P_Q, (long long)SEQ * P_Q,
             nullptr, 0, 0, nullptr, nullptr, 0, 0,
             SEQ, P_Q, E_DIM, 2, BATCH);
    };

    // ---- refinement decoder ----
    auto refine = [&](int l) {
        // cross-attention: q = rp + pos
        k_addpos<<<40, E_DIM, 0, stream>>>(rp, A(I_POS), qbuf);
        ln(qbuf, RF(l, R_CA_LNG), RF(l, R_CA_LNB), xn16, 40, E_DIM);
        gemm(xn16, E_DIM, 0, w_cawi[l], E_DIM, 0, Qb, E_DIM, 0, nullptr, 0, 0,
             RF(l, R_CA_BI), nullptr, 0, 0, 40, E_DIM, E_DIM, 0, 1);
        gemm(tp16, E_DIM, 0, w_cawi[l] + 256 * E_DIM, E_DIM, 0, Kb, E_DIM, 0,
             nullptr, 0, 0, RF(l, R_CA_BI) + 256, nullptr, 0, 0,
             TOK, E_DIM, E_DIM, 0, 1);
        gemm(tp16, E_DIM, 0, w_cawi[l] + 512 * E_DIM, E_DIM, 0, Vb, E_DIM, 0,
             nullptr, 0, 0, RF(l, R_CA_BI) + 512, nullptr, 0, 0,
             TOK, E_DIM, E_DIM, 0, 1);
        k_attn_ca<<<BATCH * H_ATT, 256, 0, stream>>>(Qb, Kb, Vb, maskp, Ob16);
        gemm(Ob16, E_DIM, 0, w_cawo[l], E_DIM, 0, rp, E_DIM, 0, nullptr, 0, 0,
             RF(l, R_CA_BO), qbuf, E_DIM, 0, 40, E_DIM, E_DIM, 0, 1);
        // self-attention
        ln(rp, RF(l, R_SA_LNG), RF(l, R_SA_LNB), xn16, 40, E_DIM);
        gemm(xn16, E_DIM, 0, w_sawi[l], E_DIM, 0, QKV, 768, 0, nullptr, 0, 0,
             RF(l, R_SA_BI), nullptr, 0, 0, 40, 768, E_DIM, 0, 1);
        k_attn_sa<<<BATCH * H_ATT, 128, 0, stream>>>(QKV, Ob16);
        gemm(Ob16, E_DIM, 0, w_sawo[l], E_DIM, 0, rp, E_DIM, 0, nullptr, 0, 0,
             RF(l, R_SA_BO), rp, E_DIM, 0, 40, E_DIM, E_DIM, 0, 1);
        // FFN
        ln(rp, RF(l, R_FFN_LNG), RF(l, R_FFN_LNB), xn16, 40, E_DIM);
        gemm(xn16, E_DIM, 0, w_fw1[l], E_DIM, 0, nullptr, 0, 0, hid40, FFN_DIM, 0,
             RF(l, R_FFN_B1), nullptr, 0, 0, 40, FFN_DIM, E_DIM, 1, 1);
        gemm(hid40, FFN_DIM, 0, w_fw2[l], FFN_DIM, 0, rp, E_DIM, 0, nullptr, 0, 0,
             RF(l, R_FFN_B2), rp, E_DIM, 0, 40, E_DIM, FFN_DIM, 0, 1);
    };

    k_proto<<<40, E_DIM, 0, stream>>>(A(I_PROTO), rp);
    make_mask(maskp);         // bias source for refine layer 0
    refine(0);
    make_mask(maskp);         // bias source for refine layer 1
    refine(1);
    make_mask(o_mask);        // final mask -> output

    // ---- final heads from rp (rows = b*10+p) ----
    auto head40 = [&](const float* lng, const float* lnb, const h16_t* w1,
                      const float* b1, const h16_t* w2, const float* b2,
                      int outdim, float* dst) {
        ln(rp, lng, lnb, lnp16, 40, E_DIM);
        gemm(lnp16, E_DIM, 0, w1, E_DIM, 0, nullptr, 0, 0, hid40, E_DIM, 0,
             b1, nullptr, 0, 0, 40, E_DIM, E_DIM, 1, 1);
        gemm(hid40, E_DIM, 0, w2, E_DIM, 0, dst, outdim, 0, nullptr, 0, 0,
             b2, nullptr, 0, 0, 40, outdim, E_DIM, 0, 1);
    };
    head40(A(I_CLS_LNG), A(I_CLS_LNB), w_cls1, A(I_CLS_B1), w_cls2, A(I_CLS_B2),
           2, logit);
    k_softmax2<<<1, 64, 0, stream>>>(logit, o_cls);
    head40(A(I_TRK_LNG), A(I_TRK_LNB), w_trk1, A(I_TRK_B1), w_trk2, A(I_TRK_B2),
           4, o_track);
    head40(A(I_PID_LNG), A(I_PID_LNB), w_pid1, A(I_PID_B1), w_pid2, A(I_PID_B2),
           6, o_pid);
}
